// AlignmentModule_19078244729283
// MI455X (gfx1250) — matrix operand order repaired and checked by reading, not compiled, not run on hardware
//
#include <hip/hip_runtime.h>
#include <hip/hip_bf16.h>
#include <math.h>

#define Bn 32
#define Sn 512
#define Tn 2048
#define Dn 256
#define Fn 80
#define FPn 96   // F padded to a multiple of 32 for WMMA K-steps

typedef __attribute__((ext_vector_type(16))) __bf16 v16bf;
typedef __attribute__((ext_vector_type(8)))  float  v8f;

union Frag { uint4 u[2]; v16bf v; };

// ---- CDNA5 async global->LDS staging (ASYNCcnt path, bypasses VGPRs) ----
__device__ __forceinline__ unsigned lds_off(const void* p) {
  // flat shared address: low 32 bits are the LDS byte address (aperture map)
  return (unsigned)(size_t)p;
}
__device__ __forceinline__ void async_ld_b128(unsigned lds, const void* gaddr) {
  asm volatile("global_load_async_to_lds_b128 %0, %1, off"
               :: "v"(lds), "v"(gaddr)
               : "memory");
}
__device__ __forceinline__ void wait_async0() {
  asm volatile("s_wait_asynccnt 0x0" ::: "memory");
}

// ---------------- elementwise prep kernels ----------------

__global__ __launch_bounds__(256) void k_cvt_bf16(const float* __restrict__ in,
                                                  __bf16* __restrict__ out, int n) {
  int i = blockIdx.x * 256 + threadIdx.x;
  if (i < n) out[i] = (__bf16)in[i];
}

__global__ __launch_bounds__(256) void k_cvt_pad_m(const float* __restrict__ in,
                                                   __bf16* __restrict__ out) {
  int idx = blockIdx.x * 256 + threadIdx.x;
  if (idx >= Bn * Tn * FPn) return;
  int i  = idx % FPn;
  int bt = idx / FPn;
  out[idx] = (i < Fn) ? (__bf16)in[bt * Fn + i] : (__bf16)0.0f;
}

// repack conv weight (O, Kin, 3) f32 -> (3, 256, Kp) bf16, zero padded in K
__global__ __launch_bounds__(256) void k_pack_w3(const float* __restrict__ in,
                                                 __bf16* __restrict__ out,
                                                 int Kin, int Kp) {
  int idx = blockIdx.x * 256 + threadIdx.x;
  if (idx >= 3 * 256 * Kp) return;
  int i = idx % Kp;
  int o = (idx / Kp) % 256;
  int j = idx / (Kp * 256);
  out[idx] = (i < Kin) ? (__bf16)in[(o * Kin + i) * 3 + j] : (__bf16)0.0f;
}

// ---------------- generic WMMA conv1d layer ----------------
// out[b, l, o] = bias[o] + sum_j sum_k in[b, l+j-pad, k] * w[j, o, k]   (opt. relu)
// Block tile: 64 positions x 256 out-channels. 8 waves in 4(m) x 2(n) grid,
// each wave owns 16 rows x 128 cols = eight 16x16 f32 accumulators.
template <int NTAPS, int KP, bool RELU>
__global__ __launch_bounds__(256) void k_conv_wmma(const __bf16* __restrict__ in,
                                                   const __bf16* __restrict__ w,
                                                   const float* __restrict__ bias,
                                                   __bf16* __restrict__ out, int L) {
  __shared__ __align__(16) __bf16 As[64 * 32];  // 4KB A tile (positions x K-slice)
  const int b    = blockIdx.y;
  const int s0   = blockIdx.x * 64;
  const int tid  = threadIdx.x;
  const int wave = tid >> 5, lane = tid & 31;
  const int wm = wave & 3;       // m sub-tile (16 rows)
  const int wn = wave >> 2;      // n half (128 cols)
  const int m_l = lane & 15, kh = lane >> 4;

  v8f acc[8];
  for (int i = 0; i < 8; ++i) acc[i] = (v8f){};

  const int pad = (NTAPS > 1) ? 1 : 0;

  for (int j = 0; j < NTAPS; ++j) {
    for (int k0 = 0; k0 < KP; k0 += 32) {
      __syncthreads();
      {  // stage A tile: 64 rows x 32 bf16 = 256 chunks of 16B (one per thread),
         // via async global->LDS; boundary rows get a zero ds_store instead.
        int row = tid >> 2, ch = tid & 3;
        int s_in = s0 + row + j - pad;
        __bf16* dst = As + row * 32 + ch * 8;
        if ((unsigned)s_in < (unsigned)L) {
          const void* g = (const void*)(in + ((size_t)b * L + s_in) * KP + k0 + ch * 8);
          async_ld_b128(lds_off(dst), g);
        } else {
          *(uint4*)dst = make_uint4(0u, 0u, 0u, 0u);
        }
      }
      wait_async0();
      __syncthreads();
      // A fragment: lane (m_l, kh) -> 16B chunks at K = kh*16 and kh*16+8
      Frag af;
      const __bf16* ap = As + (wm * 16 + m_l) * 32;
      af.u[0] = *(const uint4*)(ap + kh * 16);
      af.u[1] = *(const uint4*)(ap + kh * 16 + 8);
      // B fragments straight from packed weights: lane n=m_l holds K=kh*16..+15
      for (int nt = 0; nt < 8; ++nt) {
        int n_g = wn * 128 + nt * 16 + m_l;
        const __bf16* bp = w + ((size_t)j * 256 + n_g) * KP + k0 + kh * 16;
        Frag bfr;
        bfr.u[0] = ((const uint4*)bp)[0];
        bfr.u[1] = ((const uint4*)bp)[1];
        acc[nt] = __builtin_amdgcn_wmma_f32_16x16x32_bf16(
            false, af.v, false, bfr.v, (short)0, acc[nt], false, false);
      }
    }
  }

  // epilogue: bias (+relu), convert to bf16, store
  for (int nt = 0; nt < 8; ++nt) {
    int n_g = wn * 128 + nt * 16 + m_l;
    float bs = bias[n_g];
    for (int r = 0; r < 8; ++r) {
      int m = wm * 16 + r + 8 * kh;   // C/D layout: M = vgpr + 8*(lane/16)
      float y = acc[nt][r] + bs;
      if (RELU) y = fmaxf(y, 0.0f);
      out[((size_t)b * L + s0 + m) * Dn + n_g] = (__bf16)y;
    }
  }
}

// ---------------- row squared-norm (on bf16-rounded activations) ----------------
__global__ __launch_bounds__(256) void k_rownorm(const __bf16* __restrict__ a,
                                                 float* __restrict__ out, int rows) {
  int row  = blockIdx.x * 8 + (threadIdx.x >> 5);
  int lane = threadIdx.x & 31;
  if (row >= rows) return;
  union { uint4 u; __bf16 e[8]; } v;
  v.u = *(const uint4*)(a + (size_t)row * Dn + lane * 8);
  float s = 0.0f;
  for (int i = 0; i < 8; ++i) { float f = (float)v.e[i]; s += f * f; }
  for (int off = 16; off > 0; off >>= 1) s += __shfl_xor(s, off, 32);
  if (lane == 0) out[row] = s;
}

// ---------------- batched distance GEMM: x = mask ? -sqrt(max(hn+mn-2*hc.mc,0)) : 0
// Block tile 128(s) x 128(t), K=256. 8 waves in 4(m) x 2(n); wave = 32x64 (8 wmma/step).
__global__ __launch_bounds__(256) void k_dist_wmma(const __bf16* __restrict__ hc,
                                                   const __bf16* __restrict__ mc,
                                                   const float* __restrict__ hn,
                                                   const float* __restrict__ mn,
                                                   const int* __restrict__ mask,
                                                   float* __restrict__ out) {
  __shared__ __align__(16) __bf16 As[128 * 32];  // 8KB hc tile
  __shared__ __align__(16) __bf16 Bs[128 * 32];  // 8KB mc tile
  const int b  = blockIdx.z;
  const int s0 = blockIdx.y * 128;
  const int t0 = blockIdx.x * 128;
  const int tid  = threadIdx.x;
  const int wave = tid >> 5, lane = tid & 31;
  const int wm = wave & 3, wn = wave >> 2;
  const int m_l = lane & 15, kh = lane >> 4;

  v8f acc[2][4];
  for (int i = 0; i < 2; ++i)
    for (int j = 0; j < 4; ++j) acc[i][j] = (v8f){};

  for (int k0 = 0; k0 < Dn; k0 += 32) {
    __syncthreads();
    for (int c = tid; c < 512; c += 256) {  // A: 128 rows x 32 bf16 (async -> LDS)
      int row = c >> 2, ch = c & 3;
      async_ld_b128(lds_off(As + row * 32 + ch * 8),
                    (const void*)(hc + ((size_t)b * Sn + s0 + row) * Dn + k0 + ch * 8));
    }
    for (int c = tid; c < 512; c += 256) {  // B: 128 rows x 32 bf16 (async -> LDS)
      int row = c >> 2, ch = c & 3;
      async_ld_b128(lds_off(Bs + row * 32 + ch * 8),
                    (const void*)(mc + ((size_t)b * Tn + t0 + row) * Dn + k0 + ch * 8));
    }
    wait_async0();
    __syncthreads();

    Frag af[2], bfr[4];
    for (int mt = 0; mt < 2; ++mt) {
      const __bf16* ap = As + (wm * 32 + mt * 16 + m_l) * 32;
      af[mt].u[0] = *(const uint4*)(ap + kh * 16);
      af[mt].u[1] = *(const uint4*)(ap + kh * 16 + 8);
    }
    for (int nt = 0; nt < 4; ++nt) {
      const __bf16* bp = Bs + (wn * 64 + nt * 16 + m_l) * 32 + kh * 16;
      bfr[nt].u[0] = ((const uint4*)bp)[0];
      bfr[nt].u[1] = ((const uint4*)bp)[1];
    }
    for (int mt = 0; mt < 2; ++mt)
      for (int nt = 0; nt < 4; ++nt)
        acc[mt][nt] = __builtin_amdgcn_wmma_f32_16x16x32_bf16(
            false, af[mt].v, false, bfr[nt].v, (short)0, acc[mt][nt], false, false);
  }

  for (int mt = 0; mt < 2; ++mt)
    for (int nt = 0; nt < 4; ++nt) {
      int t = t0 + wn * 64 + nt * 16 + m_l;
      float mnv = mn[b * Tn + t];
      for (int r = 0; r < 8; ++r) {
        int s = s0 + wm * 32 + mt * 16 + r + 8 * kh;
        float sq = hn[b * Sn + s] + mnv - 2.0f * acc[mt][nt][r];
        float d = sqrtf(fmaxf(sq, 0.0f));
        float x = mask[b * Sn + s] ? -d : 0.0f;
        out[((size_t)b * Sn + s) * Tn + t] = x;
      }
    }
}

// ---------------- in-place online log-softmax over S per (b, t) column ----------------
__global__ __launch_bounds__(256) void k_logsoftmax(float* __restrict__ x) {
  int b = blockIdx.y;
  int t = blockIdx.x * 256 + threadIdx.x;
  float* col = x + (size_t)b * Sn * Tn + t;
  float mx = -INFINITY, sum = 0.0f;
  for (int s = 0; s < Sn; ++s) {
    float v = col[(size_t)s * Tn];
    if (v > mx) { sum = sum * expf(mx - v) + 1.0f; mx = v; }
    else        { sum += expf(v - mx); }
  }
  float lse = mx + logf(sum);
  for (int s = 0; s < Sn; ++s) col[(size_t)s * Tn] -= lse;
}

// ---------------- host launch ----------------
extern "C" void kernel_launch(void* const* d_in, const int* in_sizes, int n_in,
                              void* d_out, int out_size, void* d_ws, size_t ws_size,
                              hipStream_t stream) {
  const float* h   = (const float*)d_in[0];
  const float* m   = (const float*)d_in[1];
  const int*   mask= (const int*)  d_in[2];
  const float* tw1 = (const float*)d_in[3];
  const float* tb1 = (const float*)d_in[4];
  const float* tw2 = (const float*)d_in[5];
  const float* tb2 = (const float*)d_in[6];
  const float* fw1 = (const float*)d_in[7];
  const float* fb1 = (const float*)d_in[8];
  const float* fw2 = (const float*)d_in[9];
  const float* fb2 = (const float*)d_in[10];
  const float* fw3 = (const float*)d_in[11];
  const float* fb3 = (const float*)d_in[12];
  float* out = (float*)d_out;

  // workspace carve-out (256B aligned)
  size_t off = 0;
  auto alloc = [&](size_t bytes) -> char* {
    char* p = (char*)d_ws + off;
    off += (bytes + 255) & ~(size_t)255;
    return p;
  };
  __bf16* hbf = (__bf16*)alloc((size_t)Bn * Sn * Dn * 2);
  __bf16* mbf = (__bf16*)alloc((size_t)Bn * Tn * FPn * 2);
  __bf16* wt1 = (__bf16*)alloc((size_t)3 * 256 * Dn * 2);
  __bf16* wt2 = (__bf16*)alloc((size_t)256 * Dn * 2);
  __bf16* wf1 = (__bf16*)alloc((size_t)3 * 256 * FPn * 2);
  __bf16* wf2 = (__bf16*)alloc((size_t)3 * 256 * Dn * 2);
  __bf16* wf3 = (__bf16*)alloc((size_t)256 * Dn * 2);
  __bf16* a1  = (__bf16*)alloc((size_t)Bn * Sn * Dn * 2);
  __bf16* hc  = (__bf16*)alloc((size_t)Bn * Sn * Dn * 2);
  __bf16* a2  = (__bf16*)alloc((size_t)Bn * Tn * Dn * 2);
  __bf16* a3  = (__bf16*)alloc((size_t)Bn * Tn * Dn * 2);
  __bf16* mc  = (__bf16*)alloc((size_t)Bn * Tn * Dn * 2);
  float*  hnb = (float*) alloc((size_t)Bn * Sn * 4);
  float*  mnb = (float*) alloc((size_t)Bn * Tn * 4);
  (void)ws_size; (void)in_sizes; (void)n_in; (void)out_size;

  // 1) precision conversion / padding / weight repack
  {
    int n = Bn * Sn * Dn;
    k_cvt_bf16<<<(n + 255) / 256, 256, 0, stream>>>(h, hbf, n);
  }
  {
    int n = Bn * Tn * FPn;
    k_cvt_pad_m<<<(n + 255) / 256, 256, 0, stream>>>(m, mbf);
  }
  k_pack_w3<<<(3 * 256 * Dn + 255) / 256, 256, 0, stream>>>(tw1, wt1, Dn, Dn);
  k_pack_w3<<<(3 * 256 * FPn + 255) / 256, 256, 0, stream>>>(fw1, wf1, Fn, FPn);
  k_pack_w3<<<(3 * 256 * Dn + 255) / 256, 256, 0, stream>>>(fw2, wf2, Dn, Dn);
  k_cvt_bf16<<<(Dn * Dn + 255) / 256, 256, 0, stream>>>(tw2, wt2, Dn * Dn);
  k_cvt_bf16<<<(Dn * Dn + 255) / 256, 256, 0, stream>>>(fw3, wf3, Dn * Dn);

  // 2) conv stacks via WMMA
  dim3 blk(256);
  k_conv_wmma<3, Dn,  true ><<<dim3(Sn / 64, Bn), blk, 0, stream>>>(hbf, wt1, tb1, a1, Sn);
  k_conv_wmma<1, Dn,  false><<<dim3(Sn / 64, Bn), blk, 0, stream>>>(a1,  wt2, tb2, hc, Sn);
  k_conv_wmma<3, FPn, true ><<<dim3(Tn / 64, Bn), blk, 0, stream>>>(mbf, wf1, fb1, a2, Tn);
  k_conv_wmma<3, Dn,  true ><<<dim3(Tn / 64, Bn), blk, 0, stream>>>(a2,  wf2, fb2, a3, Tn);
  k_conv_wmma<1, Dn,  false><<<dim3(Tn / 64, Bn), blk, 0, stream>>>(a3,  wf3, fb3, mc, Tn);

  // 3) squared row norms of the bf16-rounded activations
  k_rownorm<<<Bn * Sn / 8, 256, 0, stream>>>(hc, hnb, Bn * Sn);
  k_rownorm<<<Bn * Tn / 8, 256, 0, stream>>>(mc, mnb, Bn * Tn);

  // 4) batched distance GEMM -> masked negative distance into d_out
  k_dist_wmma<<<dim3(Tn / 128, Sn / 128, Bn), blk, 0, stream>>>(hc, mc, hnb, mnb, mask, out);

  // 5) in-place log-softmax over S
  k_logsoftmax<<<dim3(Tn / 256, Bn), 256, 0, stream>>>(out);
}
